// SVAE_45071386805069
// MI455X (gfx1250) — matrix operand order repaired and checked by reading, not compiled, not run on hardware
//
#include <hip/hip_runtime.h>
#include <stdint.h>

// ---------------------------------------------------------------------------
// SVAE forward on gfx1250: bf16 WMMA GEMMs + fused epilogues + vMF sampling.
// ---------------------------------------------------------------------------

#define BATCH       16384
#define INPUT_DIM   1024
#define HIDDEN_DIM  2048
#define HALF_DIM    1024
#define LATENT_DIM  64
#define ROUNDS      16
#define MAX_KAPPA   6400.0f

typedef __attribute__((ext_vector_type(16))) __bf16 bf16x16;
typedef __attribute__((ext_vector_type(8)))  float  floatx8;

union Frag {
    uint4   u4[2];
    bf16x16 v;
};

__device__ __forceinline__ unsigned short f2bf(float f) {
    unsigned int x = __builtin_bit_cast(unsigned int, f);
    unsigned int r = x + 0x7FFFu + ((x >> 16) & 1u);
    return (unsigned short)(r >> 16);
}
__device__ __forceinline__ float bf2f(unsigned short h) {
    unsigned int x = ((unsigned int)h) << 16;
    return __builtin_bit_cast(float, x);
}

// -------------------------- conversion kernels ------------------------------

__global__ __launch_bounds__(256)
void convert_bf16_kernel(const float* __restrict__ in, unsigned short* __restrict__ out, int n) {
    for (int i = blockIdx.x * blockDim.x + threadIdx.x; i < n; i += gridDim.x * blockDim.x)
        out[i] = f2bf(in[i]);
}

// W is (K x N) row-major f32; Wt is (N x K) row-major bf16 (i.e. transposed).
__global__ __launch_bounds__(256)
void transpose_bf16_kernel(const float* __restrict__ W, unsigned short* __restrict__ Wt,
                           int K, int N) {
    int total = K * N;
    for (int idx = blockIdx.x * blockDim.x + threadIdx.x; idx < total;
         idx += gridDim.x * blockDim.x) {
        int n  = idx / K;
        int kk = idx - n * K;
        Wt[idx] = f2bf(W[(size_t)kk * N + n]);
    }
}

// ------------------------------ WMMA GEMM -----------------------------------
// C[M x N] = act(A[M x K](bf16, row-major) @ B[K x N] + bias)
// B supplied transposed: Bt[N x K] bf16 row-major.
// Block: 128 threads = 4 waves; wave computes 32(M) x 64(N).
// Grid: (M/128, N/64).
template<int K, int N, bool RELU, bool OUT_BF16>
__global__ __launch_bounds__(128)
void gemm_bias_act(const unsigned short* __restrict__ A,
                   const unsigned short* __restrict__ Bt,
                   const float* __restrict__ bias,
                   void* __restrict__ Out) {
    const int lane = threadIdx.x & 31;
    const int wave = threadIdx.x >> 5;
    const int half = lane >> 4;   // 0: lanes 0-15, 1: lanes 16-31
    const int l16  = lane & 15;

    const int rowBase = blockIdx.x * 128 + wave * 32;  // two 16-row m-tiles
    const int colBase = blockIdx.y * 64;               // four 16-col n-tiles

    floatx8 acc[2][4] = {};

    for (int k = 0; k < K; k += 32) {
        // A fragments: 16-bit A 16x32 layout.
        // lane half h, elem e: e<8 -> K = k + h*8 + e ; e>=8 -> K = k + 16 + h*8 + (e-8)
        Frag a[2];
#pragma unroll
        for (int i = 0; i < 2; ++i) {
            const unsigned short* ap =
                A + (size_t)(rowBase + i * 16 + l16) * K + k + half * 8;
            a[i].u4[0] = *(const uint4*)(ap);        // K offsets  h*8 + 0..7
            a[i].u4[1] = *(const uint4*)(ap + 16);   // K offsets  16 + h*8 + 0..7
        }
        // B fragments: 16-bit B 32x16 layout.
        // lane half h, col = l16, elem e: e<8 -> K = k + h*8 + e ; e>=8 -> K = k + 16 + h*8 + (e-8)
        Frag b[4];
#pragma unroll
        for (int j = 0; j < 4; ++j) {
            const unsigned short* bp =
                Bt + (size_t)(colBase + j * 16 + l16) * K + k + half * 8;
            b[j].u4[0] = *(const uint4*)(bp);
            b[j].u4[1] = *(const uint4*)(bp + 16);
        }
#pragma unroll
        for (int i = 0; i < 2; ++i)
#pragma unroll
            for (int j = 0; j < 4; ++j)
                acc[i][j] = __builtin_amdgcn_wmma_f32_16x16x32_bf16(
                    false, a[i].v, false, b[j].v, (short)0, acc[i][j], false, false);
    }

    // Epilogue. C/D layout: elem v of lane -> row = half*8 + v, col = l16.
#pragma unroll
    for (int j = 0; j < 4; ++j) {
        const int col = colBase + j * 16 + l16;
        const float bs = bias[col];
#pragma unroll
        for (int i = 0; i < 2; ++i) {
#pragma unroll
            for (int v = 0; v < 8; ++v) {
                float r = acc[i][j][v] + bs;
                if (RELU) r = fmaxf(r, 0.0f);
                const int row = rowBase + i * 16 + half * 8 + v;
                if (OUT_BF16)
                    ((unsigned short*)Out)[(size_t)row * N + col] = f2bf(r);
                else
                    ((float*)Out)[(size_t)row * N + col] = r;
            }
        }
    }
}

// ------------------------------ kappa head ----------------------------------
// kappa = clip(softplus(h2 @ Wk + bk) + 1, max=MAX_KAPPA); one wave per row.
__global__ __launch_bounds__(256)
void kappa_kernel(const unsigned short* __restrict__ h2,
                  const float* __restrict__ Wk, const float* __restrict__ bk,
                  float* __restrict__ kout) {
    const int row  = blockIdx.x * 8 + (threadIdx.x >> 5);
    const int lane = threadIdx.x & 31;
    float s = 0.0f;
    for (int kk = lane; kk < HALF_DIM; kk += 32)
        s += bf2f(h2[(size_t)row * HALF_DIM + kk]) * Wk[kk];
    for (int off = 16; off; off >>= 1) s += __shfl_xor(s, off, 32);
    if (lane == 0) {
        float v  = s + bk[0];
        float sp = (v > 20.0f) ? v : log1pf(expf(v));
        kout[row] = fminf(sp + 1.0f, MAX_KAPPA);
    }
}

// ------------------------------ vMF sampler ---------------------------------
// One wave per row; each lane owns latent components (lane, lane+32).
__global__ __launch_bounds__(256)
void sample_kernel(const float* __restrict__ mu_raw, const float* __restrict__ kappa,
                   const float* __restrict__ eps, const float* __restrict__ u,
                   const float* __restrict__ noise,
                   float* __restrict__ mu_out, unsigned short* __restrict__ z_out) {
    const int row  = blockIdx.x * 8 + (threadIdx.x >> 5);
    const int lane = threadIdx.x & 31;

    float m0 = mu_raw[(size_t)row * LATENT_DIM + lane];
    float m1 = mu_raw[(size_t)row * LATENT_DIM + lane + 32];
    float ss = m0 * m0 + m1 * m1;
    for (int off = 16; off; off >>= 1) ss += __shfl_xor(ss, off, 32);
    const float inv = 1.0f / (sqrtf(ss) + 1e-8f);
    m0 *= inv; m1 *= inv;
    mu_out[(size_t)row * LATENT_DIM + lane]      = m0;
    mu_out[(size_t)row * LATENT_DIM + lane + 32] = m1;

    const float k  = kappa[row];
    const float d  = (float)(LATENT_DIM - 1);
    const float sq4 = sqrtf(4.0f * k * k + d * d);
    const float b   = d / (sq4 + 2.0f * k);
    const float x0  = (1.0f - b) / (1.0f + b);
    const float c   = k * x0 + d * log1pf(-x0 * x0);

    float w = 0.0f, wsel = 0.0f;
    bool found = false;
    for (int t = 0; t < ROUNDS; ++t) {
        const float e  = eps[(size_t)t * BATCH + row];
        const float uu = u[(size_t)t * BATCH + row];
        w = (1.0f - (1.0f + b) * e) / (1.0f - (1.0f - b) * e);
        const float tv = k * w + d * log1pf(-x0 * w) - c;
        const bool acc = tv >= logf(fmaxf(uu, 1e-12f));
        if (acc && !found) { wsel = w; found = true; }
    }
    const float wf = found ? wsel : w;

    float n0 = noise[(size_t)row * LATENT_DIM + lane];
    float n1 = noise[(size_t)row * LATENT_DIM + lane + 32];
    float dp = n0 * m0 + n1 * m1;
    for (int off = 16; off; off >>= 1) dp += __shfl_xor(dp, off, 32);
    float p0 = n0 - dp * m0, p1 = n1 - dp * m1;
    float pn = p0 * p0 + p1 * p1;
    for (int off = 16; off; off >>= 1) pn += __shfl_xor(pn, off, 32);
    const float invp = 1.0f / (sqrtf(pn) + 1e-8f);
    const float tang = sqrtf(fmaxf(1.0f - wf * wf, 0.0f));
    z_out[(size_t)row * LATENT_DIM + lane]      = f2bf(wf * m0 + tang * p0 * invp);
    z_out[(size_t)row * LATENT_DIM + lane + 32] = f2bf(wf * m1 + tang * p1 * invp);
}

// ------------------------------ launcher ------------------------------------

static constexpr size_t alignup(size_t x) { return (x + 255) & ~size_t(255); }

extern "C" void kernel_launch(void* const* d_in, const int* in_sizes, int n_in,
                              void* d_out, int out_size, void* d_ws, size_t ws_size,
                              hipStream_t stream) {
    const float* x    = (const float*)d_in[0];
    const float* W1   = (const float*)d_in[1];
    const float* b1   = (const float*)d_in[2];
    const float* W2   = (const float*)d_in[3];
    const float* b2   = (const float*)d_in[4];
    const float* Wmu  = (const float*)d_in[5];
    const float* bmu  = (const float*)d_in[6];
    const float* Wk   = (const float*)d_in[7];
    const float* bk   = (const float*)d_in[8];
    const float* W3   = (const float*)d_in[9];
    const float* b3   = (const float*)d_in[10];
    const float* W4   = (const float*)d_in[11];
    const float* b4   = (const float*)d_in[12];
    const float* W5   = (const float*)d_in[13];
    const float* b5   = (const float*)d_in[14];
    const float* eps  = (const float*)d_in[15];
    const float* u    = (const float*)d_in[16];
    const float* noise= (const float*)d_in[17];

    float* out = (float*)d_out;
    float* xr_out    = out;                                      // 16384 x 1024
    float* mu_out    = out + (size_t)BATCH * INPUT_DIM;          // 16384 x 64
    float* kappa_out = mu_out + (size_t)BATCH * LATENT_DIM;      // 16384 x 1

    // Workspace layout (bytes). mu_raw and z alias the x_bf16 region
    // (x_bf16 is dead after GEMM1); h1 aliases hd2, h2 aliases hd1.
    constexpr size_t SZ_XB   = (size_t)BATCH * INPUT_DIM * 2;    // 33.5 MB
    constexpr size_t SZ_W1T  = (size_t)INPUT_DIM * HIDDEN_DIM * 2;
    constexpr size_t SZ_W2T  = (size_t)HIDDEN_DIM * HALF_DIM * 2;
    constexpr size_t SZ_WMUT = (size_t)HALF_DIM * LATENT_DIM * 2;
    constexpr size_t SZ_W3T  = (size_t)LATENT_DIM * HALF_DIM * 2;
    constexpr size_t SZ_W4T  = (size_t)HALF_DIM * HIDDEN_DIM * 2;
    constexpr size_t SZ_W5T  = (size_t)HIDDEN_DIM * INPUT_DIM * 2;
    constexpr size_t SZ_H1   = (size_t)BATCH * HIDDEN_DIM * 2;   // also hd2
    constexpr size_t SZ_H2   = (size_t)BATCH * HALF_DIM * 2;     // also hd1
    constexpr size_t SZ_MUR  = (size_t)BATCH * LATENT_DIM * 4;

    constexpr size_t OFF_XB   = 0;
    constexpr size_t OFF_MUR  = 0;                       // alias into XB region
    constexpr size_t OFF_Z    = alignup(SZ_MUR);         // alias into XB region
    constexpr size_t OFF_W1T  = alignup(SZ_XB);
    constexpr size_t OFF_W2T  = OFF_W1T + alignup(SZ_W1T);
    constexpr size_t OFF_WMUT = OFF_W2T + alignup(SZ_W2T);
    constexpr size_t OFF_W3T  = OFF_WMUT + alignup(SZ_WMUT);
    constexpr size_t OFF_W4T  = OFF_W3T + alignup(SZ_W3T);
    constexpr size_t OFF_W5T  = OFF_W4T + alignup(SZ_W4T);
    constexpr size_t OFF_H1   = OFF_W5T + alignup(SZ_W5T);   // also hd2
    constexpr size_t OFF_H2   = OFF_H1 + alignup(SZ_H1);     // also hd1

    char* ws = (char*)d_ws;
    unsigned short* xb   = (unsigned short*)(ws + OFF_XB);
    unsigned short* W1t  = (unsigned short*)(ws + OFF_W1T);
    unsigned short* W2t  = (unsigned short*)(ws + OFF_W2T);
    unsigned short* Wmut = (unsigned short*)(ws + OFF_WMUT);
    unsigned short* W3t  = (unsigned short*)(ws + OFF_W3T);
    unsigned short* W4t  = (unsigned short*)(ws + OFF_W4T);
    unsigned short* W5t  = (unsigned short*)(ws + OFF_W5T);
    unsigned short* h1   = (unsigned short*)(ws + OFF_H1);
    unsigned short* h2   = (unsigned short*)(ws + OFF_H2);
    unsigned short* hd1  = h2;   // alias
    unsigned short* hd2  = h1;   // alias
    float*          mur  = (float*)(ws + OFF_MUR);
    unsigned short* zb   = (unsigned short*)(ws + OFF_Z);

    const dim3 blkC(256);

    // 1) convert x and weights to bf16 (weights transposed to N-major).
    convert_bf16_kernel<<<4096, blkC, 0, stream>>>(x, xb, BATCH * INPUT_DIM);
    transpose_bf16_kernel<<<2048, blkC, 0, stream>>>(W1, W1t, INPUT_DIM, HIDDEN_DIM);
    transpose_bf16_kernel<<<2048, blkC, 0, stream>>>(W2, W2t, HIDDEN_DIM, HALF_DIM);
    transpose_bf16_kernel<<<512,  blkC, 0, stream>>>(Wmu, Wmut, HALF_DIM, LATENT_DIM);
    transpose_bf16_kernel<<<512,  blkC, 0, stream>>>(W3, W3t, LATENT_DIM, HALF_DIM);
    transpose_bf16_kernel<<<2048, blkC, 0, stream>>>(W4, W4t, HALF_DIM, HIDDEN_DIM);
    transpose_bf16_kernel<<<2048, blkC, 0, stream>>>(W5, W5t, HIDDEN_DIM, INPUT_DIM);

    // 2) encoder GEMMs.
    gemm_bias_act<INPUT_DIM, HIDDEN_DIM, true, true>
        <<<dim3(BATCH / 128, HIDDEN_DIM / 64), 128, 0, stream>>>(xb, W1t, b1, h1);
    gemm_bias_act<HIDDEN_DIM, HALF_DIM, true, true>
        <<<dim3(BATCH / 128, HALF_DIM / 64), 128, 0, stream>>>(h1, W2t, b2, h2);
    gemm_bias_act<HALF_DIM, LATENT_DIM, false, false>
        <<<dim3(BATCH / 128, 1), 128, 0, stream>>>(h2, Wmut, bmu, mur);

    // 3) kappa head + vMF sampling.
    kappa_kernel<<<BATCH / 8, blkC, 0, stream>>>(h2, Wk, bk, kappa_out);
    sample_kernel<<<BATCH / 8, blkC, 0, stream>>>(mur, kappa_out, eps, u, noise,
                                                  mu_out, zb);

    // 4) decoder GEMMs.
    gemm_bias_act<LATENT_DIM, HALF_DIM, true, true>
        <<<dim3(BATCH / 128, HALF_DIM / 64), 128, 0, stream>>>(zb, W3t, b3, hd1);
    gemm_bias_act<HALF_DIM, HIDDEN_DIM, true, true>
        <<<dim3(BATCH / 128, HIDDEN_DIM / 64), 128, 0, stream>>>(hd1, W4t, b4, hd2);
    gemm_bias_act<HIDDEN_DIM, INPUT_DIM, false, false>
        <<<dim3(BATCH / 128, INPUT_DIM / 64), 128, 0, stream>>>(hd2, W5t, b5, xr_out);
}
